// ConvSelfAttention_9148280341193
// MI455X (gfx1250) — matrix operand order repaired and checked by reading, not compiled, not run on hardware
//
#include <hip/hip_runtime.h>
#include <hip/hip_bf16.h>

// ---------------------------------------------------------------------------
// CDNA5 (gfx1250) bf16 WMMA types
// ---------------------------------------------------------------------------
typedef __attribute__((ext_vector_type(16))) __bf16 v16bf;
typedef __attribute__((ext_vector_type(8)))  float  v8f;

union FragAB {
    v16bf        v;
    unsigned int u[8];
};

__device__ __forceinline__ unsigned short f2bf(float f) {
    // round-to-nearest-even f32 -> bf16
    unsigned int u = __float_as_uint(f);
    u += 0x7FFFu + ((u >> 16) & 1u);
    return (unsigned short)(u >> 16);
}

// ---------------------------------------------------------------------------
// Conversion kernels
// ---------------------------------------------------------------------------
__global__ void cvt_bf16(const float* __restrict__ src,
                         unsigned short* __restrict__ dst, int n) {
    int i = blockIdx.x * 256 + threadIdx.x;
    if (i < n) dst[i] = f2bf(src[i]);
}

// x [512][4096] f32  ->  xT [4096][512] bf16
__global__ void cvt_xT(const float* __restrict__ x,
                       unsigned short* __restrict__ xT) {
    int i = blockIdx.x * 256 + threadIdx.x;   // 512*4096 elements
    int c = i >> 12;
    int p = i & 4095;
    xT[p * 512 + c] = f2bf(x[i]);
}

// ---------------------------------------------------------------------------
// Projection GEMM:  D[M][4096] = A[M][K] * B[K][4096],  B[k][n] = BT[n][k]
// One 16x16 output tile per wave, bf16 WMMA, f32 accumulate.
// transposeOut=1 stores Out[n][m] (row stride M), else Out[m][n] (stride 4096)
// ---------------------------------------------------------------------------
__global__ __launch_bounds__(256) void proj_gemm(
    const unsigned short* __restrict__ A,   // [M][K] bf16 row-major
    const unsigned short* __restrict__ BT,  // [4096][K] bf16 row-major
    unsigned short* __restrict__ Out,
    int M, int K, int transposeOut)
{
    const int lane = threadIdx.x & 31;
    const int wave = threadIdx.x >> 5;
    const int tile = blockIdx.x * 8 + wave;
    const int tilesN = 4096 / 16;
    const int row0 = (tile / tilesN) * 16;
    const int col0 = (tile % tilesN) * 16;
    const int m = lane & 15;
    const int h = lane >> 4;

    v8f acc = {0.f, 0.f, 0.f, 0.f, 0.f, 0.f, 0.f, 0.f};
    const unsigned short* aRow = A  + (row0 + m) * K;
    const unsigned short* bRow = BT + (col0 + m) * K;

    for (int kk = 0; kk < K; kk += 32) {
        FragAB a, b;
#pragma unroll
        for (int v = 0; v < 8; ++v) {
            int ka = kk + 16 * h + 2 * v;
            int kb = kk + 16 * h + 2 * v;
            a.u[v] = *(const unsigned int*)(aRow + ka);
            b.u[v] = *(const unsigned int*)(bRow + kb);
        }
        acc = __builtin_amdgcn_wmma_f32_16x16x32_bf16(
            false, a.v, false, b.v, (short)0, acc, false, false);
    }

#pragma unroll
    for (int r = 0; r < 8; ++r) {
        int orow = row0 + r + 8 * h;
        int ocol = col0 + m;
        unsigned short val = f2bf(acc[r]);
        if (transposeOut) Out[ocol * M + orow] = val;
        else              Out[orow * 4096 + ocol] = val;
    }
}

// ---------------------------------------------------------------------------
// Fused flash-attention:
//   S = kxT * qxT^T (K=64), online softmax over j, O += P * vx^T, out = O / l
// Block: 256 threads (8 waves), 32 query rows per block, key blocks of 64.
// ---------------------------------------------------------------------------
#define BM 32
#define BN 64

__global__ __launch_bounds__(256) void flash_attn(
    const unsigned short* __restrict__ kxT,  // [4096][64] bf16
    const unsigned short* __restrict__ qxT,  // [4096][64] bf16
    const unsigned short* __restrict__ vxb,  // [512][4096] bf16
    float* __restrict__ out)                 // [4096][512] f32
{
    __shared__ float          Sld[BM][BN + 1];   // padded: no bank conflicts
    __shared__ unsigned short Pld[BM][BN + 4];   // 68 -> dword-aligned rows
    __shared__ float mRow[BM], lRow[BM], aRow[BM];

    const int tid  = threadIdx.x;
    const int lane = tid & 31;
    const int wave = tid >> 5;
    const int m = lane & 15;
    const int h = lane >> 4;
    const int row0 = blockIdx.x * BM;

    // S-phase tile for this wave (2x4 tiles of 16x16 covering 32x64)
    const int ti = wave >> 2;          // 0..1
    const int tj = wave & 3;           // 0..3
    // O-phase slice: 16 rows x 128 cols per wave
    const int R  = (wave & 1) * 16;
    const int C0 = (wave >> 1) * 128;

    if (tid < BM) { mRow[tid] = -3.402823466e38f; lRow[tid] = 0.f; }
    __syncthreads();

    const v8f vzero = {0.f, 0.f, 0.f, 0.f, 0.f, 0.f, 0.f, 0.f};
    v8f Oacc[8];
#pragma unroll
    for (int ct = 0; ct < 8; ++ct) Oacc[ct] = vzero;

    const unsigned short* aK = kxT + (row0 + ti * 16 + m) * 64;

    for (int jb = 0; jb < 64; ++jb) {
        const int j0 = jb * 64;

        // ---- phase 1: S tile = kxT-rows x qxT-rows ----
        {
            const unsigned short* bQ = qxT + (j0 + tj * 16 + m) * 64;
            v8f s = vzero;
#pragma unroll
            for (int kc = 0; kc < 64; kc += 32) {
                FragAB a, b;
#pragma unroll
                for (int v = 0; v < 8; ++v) {
                    a.u[v] = *(const unsigned int*)(aK + kc + 16 * h + 2 * v);
                    b.u[v] = *(const unsigned int*)(bQ + kc + 16 * h + 2 * v);
                }
                s = __builtin_amdgcn_wmma_f32_16x16x32_bf16(
                    false, a.v, false, b.v, (short)0, s, false, false);
            }
#pragma unroll
            for (int r = 0; r < 8; ++r)
                Sld[ti * 16 + r + 8 * h][tj * 16 + m] = s[r];

            // prefetch next j-block's vx fragments into cache while we
            // softmax (lowers to global_prefetch_b8)
            if (jb < 63) {
#pragma unroll
                for (int ct = 0; ct < 8; ++ct)
                    __builtin_prefetch(vxb + (C0 + ct * 16 + m) * 4096 + j0 + 64, 0, 3);
            }
        }
        __syncthreads();

        // ---- phase 2: online softmax, 8 threads per row (all 256 active) ----
        {
            const int row = tid >> 3;      // 0..31
            const int seg = tid & 7;       // 0..7, 8 elements each
            float e0[8];
            float bm = -3.402823466e38f;
#pragma unroll
            for (int e = 0; e < 8; ++e) {
                e0[e] = Sld[row][seg * 8 + e];
                bm = fmaxf(bm, e0[e]);
            }
            // max-reduce across the 8 threads of this row (same wave)
            bm = fmaxf(bm, __shfl_xor(bm, 1));
            bm = fmaxf(bm, __shfl_xor(bm, 2));
            bm = fmaxf(bm, __shfl_xor(bm, 4));
            float mo = mRow[row];
            float mn = fmaxf(mo, bm);
            float sum = 0.f;
#pragma unroll
            for (int e = 0; e < 8; ++e) {
                float ex = __expf(e0[e] - mn);
                sum += ex;
                Pld[row][seg * 8 + e] = f2bf(ex);
            }
            sum += __shfl_xor(sum, 1);
            sum += __shfl_xor(sum, 2);
            sum += __shfl_xor(sum, 4);
            if (seg == 0) {
                float al = __expf(mo - mn);
                lRow[row] = lRow[row] * al + sum;
                mRow[row] = mn;
                aRow[row] = al;
            }
        }
        __syncthreads();

        // ---- phase 3: rescale O and accumulate P * vx^T ----
        {
            float al[8];
#pragma unroll
            for (int r = 0; r < 8; ++r) al[r] = aRow[R + r + 8 * h];

            FragAB a0, a1;
#pragma unroll
            for (int v = 0; v < 8; ++v) {
                int ka = 16 * h + 2 * v;
                a0.u[v] = *(const unsigned int*)(&Pld[R + m][ka]);
                a1.u[v] = *(const unsigned int*)(&Pld[R + m][32 + ka]);
            }
#pragma unroll
            for (int ct = 0; ct < 8; ++ct) {
                const unsigned short* bV = vxb + (C0 + ct * 16 + m) * 4096 + j0;
                FragAB b0, b1;
#pragma unroll
                for (int v = 0; v < 8; ++v) {
                    int kb = 16 * h + 2 * v;
                    b0.u[v] = *(const unsigned int*)(bV + kb);
                    b1.u[v] = *(const unsigned int*)(bV + 32 + kb);
                }
                v8f t = Oacc[ct];
#pragma unroll
                for (int e = 0; e < 8; ++e) t[e] *= al[e];
                t = __builtin_amdgcn_wmma_f32_16x16x32_bf16(
                    false, a0.v, false, b0.v, (short)0, t, false, false);
                t = __builtin_amdgcn_wmma_f32_16x16x32_bf16(
                    false, a1.v, false, b1.v, (short)0, t, false, false);
                Oacc[ct] = t;
            }
        }
        __syncthreads();
    }

    // ---- finalize: divide by row sums, store f32 [N][C] ----
    float linv[8];
#pragma unroll
    for (int r = 0; r < 8; ++r) linv[r] = 1.f / lRow[R + r + 8 * h];
#pragma unroll
    for (int ct = 0; ct < 8; ++ct) {
#pragma unroll
        for (int r = 0; r < 8; ++r) {
            out[(row0 + R + r + 8 * h) * 512 + C0 + ct * 16 + m] =
                Oacc[ct][r] * linv[r];
        }
    }
}

// ---------------------------------------------------------------------------
// Host launcher
// ---------------------------------------------------------------------------
extern "C" void kernel_launch(void* const* d_in, const int* in_sizes, int n_in,
                              void* d_out, int out_size, void* d_ws, size_t ws_size,
                              hipStream_t stream) {
    const float* x  = (const float*)d_in[0];  // [512][4096]
    const float* Wk = (const float*)d_in[1];  // [64][512]
    const float* Wq = (const float*)d_in[2];  // [64][512]
    const float* Wv = (const float*)d_in[3];  // [512][512]
    float* out = (float*)d_out;               // [4096][512]

    char* ws = (char*)d_ws;
    unsigned short* xT  = (unsigned short*)ws; ws += (size_t)4096 * 512 * 2;
    unsigned short* Wkb = (unsigned short*)ws; ws += (size_t)64 * 512 * 2;
    unsigned short* Wqb = (unsigned short*)ws; ws += (size_t)64 * 512 * 2;
    unsigned short* Wvb = (unsigned short*)ws; ws += (size_t)512 * 512 * 2;
    unsigned short* kxT = (unsigned short*)ws; ws += (size_t)4096 * 64 * 2;
    unsigned short* qxT = (unsigned short*)ws; ws += (size_t)4096 * 64 * 2;
    unsigned short* vxb = (unsigned short*)ws; ws += (size_t)512 * 4096 * 2;
    (void)in_sizes; (void)n_in; (void)out_size; (void)ws_size;

    // bf16 conversions
    cvt_xT<<<(512 * 4096) / 256, 256, 0, stream>>>(x, xT);
    cvt_bf16<<<(64 * 512 + 255) / 256, 256, 0, stream>>>(Wk, Wkb, 64 * 512);
    cvt_bf16<<<(64 * 512 + 255) / 256, 256, 0, stream>>>(Wq, Wqb, 64 * 512);
    cvt_bf16<<<(512 * 512 + 255) / 256, 256, 0, stream>>>(Wv, Wvb, 512 * 512);

    // projections (WMMA GEMMs)
    // kx, qx: [64][4096], stored transposed as [4096][64]
    proj_gemm<<<(4 * 256) / 8, 256, 0, stream>>>(Wkb, xT, kxT, 64, 512, 1);
    proj_gemm<<<(4 * 256) / 8, 256, 0, stream>>>(Wqb, xT, qxT, 64, 512, 1);
    // vx: [512][4096], kept row-major
    proj_gemm<<<(32 * 256) / 8, 256, 0, stream>>>(Wvb, xT, vxb, 512, 512, 0);

    // fused flash attention: 4096/32 = 128 blocks
    flash_attn<<<128, 256, 0, stream>>>(kxT, qxT, vxb, out);
}
